// EncoderLayer_73315091743398
// MI455X (gfx1250) — matrix operand order repaired and checked by reading, not compiled, not run on hardware
//
#include <hip/hip_runtime.h>
#include <hip/hip_bf16.h>

typedef __attribute__((ext_vector_type(16))) __bf16       bf16x16;
typedef __attribute__((ext_vector_type(8)))  float        f32x8;
typedef __attribute__((ext_vector_type(4)))  float        f32x4;
typedef __attribute__((ext_vector_type(4)))  unsigned int u32x4;

union FragU { bf16x16 v; u32x4 q[2]; };

__device__ __forceinline__ unsigned short bf16_of(float x) {
  unsigned u = __float_as_uint(x);
  return (unsigned short)((u + 0x7FFFu + ((u >> 16) & 1u)) >> 16);  // RNE truncate
}

enum { EP_NONE = 0, EP_PE = 1, EP_RELU_BIAS_BF16 = 2, EP_BIAS = 3 };

#define LDSS 40  // padded bf16 row stride (80B): 16B-aligned chunks, spread banks

// ---------------------------------------------------------------------------
// Vectorized f32 -> bf16 (RNE) pack: 8 elems/thread, b128 in / b128 out.
// n must be a multiple of 2048 (all our operands are).
__global__ __launch_bounds__(256) void cvt_bf16_kernel(const float* __restrict__ in,
                                                       unsigned short* __restrict__ out,
                                                       int n8) {
  const int i = blockIdx.x * 256 + threadIdx.x;
  if (i >= n8) return;
  const f32x4 a = *(const f32x4*)(in + (size_t)i * 8);
  const f32x4 b = *(const f32x4*)(in + (size_t)i * 8 + 4);
  u32x4 p;
  p.x = (unsigned)bf16_of(a.x) | ((unsigned)bf16_of(a.y) << 16);
  p.y = (unsigned)bf16_of(a.z) | ((unsigned)bf16_of(a.w) << 16);
  p.z = (unsigned)bf16_of(b.x) | ((unsigned)bf16_of(b.y) << 16);
  p.w = (unsigned)bf16_of(b.z) | ((unsigned)bf16_of(b.w) << 16);
  *(u32x4*)(out + (size_t)i * 8) = p;
}

// ---------------------------------------------------------------------------
// C[M,N] = A[M,K] @ B[N,K]^T, A/B bf16 row-major (K contiguous), f32 accum.
// Block 128x128, 8 waves, 64x32 per wave (4x2 WMMA tiles), K-step 32.
template <int EP>
__global__ __launch_bounds__(256) void wmma_gemm_kernel(
    const unsigned short* __restrict__ A, const unsigned short* __restrict__ B,
    const float* __restrict__ bias, void* __restrict__ Cout,
    int M, int N, int K) {
  __shared__ unsigned short sA[128 * LDSS];
  __shared__ unsigned short sB[128 * LDSS];

  const int tid  = threadIdx.x;
  const int lane = tid & 31;
  const int wave = tid >> 5;
  const int wm = (wave >> 2) * 64;   // 2 waves tile M
  const int wn = (wave & 3) * 32;    // 4 waves tile N
  const int rowBase = blockIdx.x * 128;
  const int colBase = blockIdx.y * 128;

  f32x8 acc[4][2] = {};

  for (int k0 = 0; k0 < K; k0 += 32) {
    __syncthreads();
    // Stage 128x32 bf16 tiles: pure b128 global -> b128 LDS copies.
    // 512 chunks of 8 bf16; thread handles chunks tid and tid+256.
#pragma unroll
    for (int it = 0; it < 2; ++it) {
      const int c = tid + it * 256;
      const int r = c >> 2, col = (c & 3) << 3;
      *(u32x4*)&sA[r * LDSS + col] =
          *(const u32x4*)&A[(size_t)(rowBase + r) * K + k0 + col];
      *(u32x4*)&sB[r * LDSS + col] =
          *(const u32x4*)&B[(size_t)(colBase + r) * K + k0 + col];
    }
    if (k0 + 32 < K) {  // pull next K-tile toward the caches (global_prefetch_b8)
      __builtin_prefetch(&A[(size_t)(rowBase + (tid & 127)) * K + k0 + 32], 0, 3);
      __builtin_prefetch(&B[(size_t)(colBase + (tid & 127)) * K + k0 + 32], 0, 3);
    }
    __syncthreads();

    // A fragment (16x32 MxK): lane<16 -> K{0..7,16..23}, lane>=16 -> K{8..15,24..31}
    bf16x16 afr[4], bfr[2];
    const int ar = lane & 15;
    const int ak = (lane >> 4) << 3;   // 0 or 8
    const int bk = (lane >> 4) << 3;   // 0 or 8 : lane<16 K{0..7,16..23}, lane>=16 K{8..15,24..31}
#pragma unroll
    for (int mf = 0; mf < 4; ++mf) {
      const unsigned short* p = &sA[(wm + mf * 16 + ar) * LDSS + ak];
      FragU f;
      f.q[0] = *(const u32x4*)(p);        // K = ak .. ak+7
      f.q[1] = *(const u32x4*)(p + 16);   // K = ak+16 .. ak+23
      afr[mf] = f.v;
    }
#pragma unroll
    for (int nf = 0; nf < 2; ++nf) {
      const unsigned short* p = &sB[(wn + nf * 16 + ar) * LDSS + bk];
      FragU f;
      f.q[0] = *(const u32x4*)(p);        // K = bk .. bk+7
      f.q[1] = *(const u32x4*)(p + 16);   // K = bk+16 .. bk+23
      bfr[nf] = f.v;
    }
#pragma unroll
    for (int mf = 0; mf < 4; ++mf)
#pragma unroll
      for (int nf = 0; nf < 2; ++nf)
        acc[mf][nf] = __builtin_amdgcn_wmma_f32_16x16x32_bf16(
            false, afr[mf], false, bfr[nf], (short)0, acc[mf][nf], false, false);
  }

  // C/D layout: VGPR v, lane L -> N = L%16, M = v + 8*(L/16)
  constexpr float PEC = -0.03597778f;  // -ln(10000)/256
#pragma unroll
  for (int mf = 0; mf < 4; ++mf) {
#pragma unroll
    for (int nf = 0; nf < 2; ++nf) {
      const int n = colBase + wn + nf * 16 + (lane & 15);
      const int m0 = rowBase + wm + mf * 16 + ((lane >> 4) << 3);
#pragma unroll
      for (int v = 0; v < 8; ++v) {
        const int m = m0 + v;
        float val = acc[mf][nf][v];
        if (EP == EP_PE) {
          const int s = m >> 4;  // row m = s*16 + h*4 + w
          const float ang = (float)s * __expf((float)(n >> 1) * PEC);
          val += (n & 1) ? __cosf(ang) : __sinf(ang);
          ((float*)Cout)[(size_t)m * N + n] = val;
        } else if (EP == EP_RELU_BIAS_BF16) {
          val = fmaxf(val + bias[n], 0.0f);
          ((unsigned short*)Cout)[(size_t)m * N + n] = bf16_of(val);
        } else if (EP == EP_BIAS) {
          ((float*)Cout)[(size_t)m * N + n] = val + bias[n];
        } else {
          ((float*)Cout)[(size_t)m * N + n] = val;
        }
      }
    }
  }
}

// ---------------------------------------------------------------------------
// Wfc_fold[n][k] = sum_{rep<8} Wfc[n][rep*64+k]   (512x64)
__global__ __launch_bounds__(256) void fold_wfc_kernel(const float* __restrict__ Wfc,
                                                       float* __restrict__ Wfold) {
  const int idx = blockIdx.x * 256 + threadIdx.x;
  if (idx >= 512 * 64) return;
  const int n = idx >> 6, k = idx & 63;
  float s = 0.f;
#pragma unroll
  for (int r = 0; r < 8; ++r) s += Wfc[(size_t)n * 512 + r * 64 + k];
  Wfold[idx] = s;
}

// ---------------------------------------------------------------------------
// Per (group g, head i): r[n]=rowsum_64(Q), a=softmax(r), V[b]=sum_n a[n]*Qrow_n
__global__ __launch_bounds__(256) void attn_reduce_kernel(const float* __restrict__ Q,
                                                          float* __restrict__ V) {
  const int b = blockIdx.x;            // 0..127 = g*8 + head
  const int g = b >> 3, head = b & 7;
  const float* base = Q + (size_t)g * 512 * 512 + head * 64;
  __shared__ float r[512];
  __shared__ float red[8];
  __shared__ float pbuf[256];
  const int tid = threadIdx.x;

  for (int n = tid; n < 512; n += 256) {
    const float* row = base + (size_t)n * 512;
    float s = 0.f;
#pragma unroll 8
    for (int k = 0; k < 64; ++k) s += row[k];
    r[n] = s;
  }
  __syncthreads();

  float mx = -3.4e38f;
  for (int n = tid; n < 512; n += 256) mx = fmaxf(mx, r[n]);
  for (int o = 16; o; o >>= 1) mx = fmaxf(mx, __shfl_xor(mx, o, 32));
  if ((tid & 31) == 0) red[tid >> 5] = mx;
  __syncthreads();
  mx = red[0];
#pragma unroll
  for (int w = 1; w < 8; ++w) mx = fmaxf(mx, red[w]);
  __syncthreads();

  float ssum = 0.f;
  for (int n = tid; n < 512; n += 256) {
    const float e = __expf(r[n] - mx);
    r[n] = e;
    ssum += e;
  }
  for (int o = 16; o; o >>= 1) ssum += __shfl_xor(ssum, o, 32);
  if ((tid & 31) == 0) red[tid >> 5] = ssum;
  __syncthreads();
  float tot = 0.f;
#pragma unroll
  for (int w = 0; w < 8; ++w) tot += red[w];
  const float inv = 1.0f / tot;
  __syncthreads();

  const int k = tid & 63, seg = tid >> 6;
  const float* segbase = base + (size_t)(seg * 128) * 512;
  float part = 0.f;
  for (int n = 0; n < 128; ++n) part += r[seg * 128 + n] * segbase[(size_t)n * 512 + k];
  pbuf[tid] = part;
  __syncthreads();
  if (tid < 64)
    V[b * 64 + tid] =
        (pbuf[tid] + pbuf[tid + 64] + pbuf[tid + 128] + pbuf[tid + 192]) * inv;
}

// ---------------------------------------------------------------------------
// Out[m] = LN(X[m] + Add[BCAST ? m>>6 : m]) * g + b ; one wave per 512-wide row.
// Optionally also emits a bf16 copy (for feeding the next WMMA GEMM).
template <bool BCAST, bool EMIT_BF16>
__global__ __launch_bounds__(256) void ln_kernel(const float* __restrict__ X,
                                                 const float* __restrict__ Add,
                                                 const float* __restrict__ g,
                                                 const float* __restrict__ b,
                                                 float* __restrict__ Out,
                                                 unsigned short* __restrict__ OutBf) {
  const int wave = threadIdx.x >> 5, lane = threadIdx.x & 31;
  const int m = blockIdx.x * 8 + wave;
  const float* xr = X + (size_t)m * 512;
  const float* ar = Add + (size_t)(BCAST ? (m >> 6) : m) * 512;
  float vals[16];
  float s = 0.f;
#pragma unroll
  for (int i = 0; i < 16; ++i) {
    const int c = lane + i * 32;
    const float t = xr[c] + ar[c];
    vals[i] = t;
    s += t;
  }
  for (int o = 16; o; o >>= 1) s += __shfl_xor(s, o, 32);
  const float mean = s * (1.0f / 512.0f);
  float vs = 0.f;
#pragma unroll
  for (int i = 0; i < 16; ++i) {
    const float d = vals[i] - mean;
    vs += d * d;
  }
  for (int o = 16; o; o >>= 1) vs += __shfl_xor(vs, o, 32);
  const float rstd = rsqrtf(vs * (1.0f / 512.0f) + 1e-5f);
#pragma unroll
  for (int i = 0; i < 16; ++i) {
    const int c = lane + i * 32;
    const float y = (vals[i] - mean) * rstd * g[c] + b[c];
    Out[(size_t)m * 512 + c] = y;
    if (EMIT_BF16) OutBf[(size_t)m * 512 + c] = bf16_of(y);
  }
}

// ---------------------------------------------------------------------------
extern "C" void kernel_launch(void* const* d_in, const int* in_sizes, int n_in,
                              void* d_out, int out_size, void* d_ws, size_t ws_size,
                              hipStream_t stream) {
  (void)in_sizes; (void)n_in; (void)out_size; (void)ws_size;
  const float* x   = (const float*)d_in[0];
  const float* Wq  = (const float*)d_in[1];
  const float* Wfc = (const float*)d_in[2];
  const float* W1  = (const float*)d_in[3];
  const float* b1  = (const float*)d_in[4];
  const float* W2  = (const float*)d_in[5];
  const float* b2  = (const float*)d_in[6];
  const float* g1  = (const float*)d_in[7];
  const float* be1 = (const float*)d_in[8];
  const float* g2  = (const float*)d_in[9];
  const float* be2 = (const float*)d_in[10];

  const int M = 8192, D = 512, DFF = 2048;
  typedef unsigned short us;

  char* w = (char*)d_ws;
  auto alloc = [&](size_t bytes) {
    void* p = (void*)w;
    w += (bytes + 255) & ~(size_t)255;
    return p;
  };
  float* Q      = (float*)alloc((size_t)M * D * 4);     // q = x@Wq^T + PE (f32)
  float* Vr     = (float*)alloc((size_t)128 * 64 * 4);  // context vectors
  float* Wfold  = (float*)alloc((size_t)D * 64 * 4);    // folded Wfc
  float* AO     = (float*)alloc((size_t)128 * D * 4);   // attn_out rows (b = m>>6)
  float* O1     = (float*)alloc((size_t)M * D * 4);     // LN1 output f32
  float* Ff     = (float*)alloc((size_t)M * D * 4);     // FFN output (pre-LN2)
  us* Xbf     = (us*)alloc((size_t)M * D * 2);
  us* Wqbf    = (us*)alloc((size_t)D * D * 2);
  us* W1bf    = (us*)alloc((size_t)DFF * D * 2);
  us* W2bf    = (us*)alloc((size_t)D * DFF * 2);
  us* O1bf    = (us*)alloc((size_t)M * D * 2);
  us* Hff     = (us*)alloc((size_t)M * DFF * 2);        // relu hidden, bf16
  us* Vbf     = (us*)alloc((size_t)128 * 64 * 2);
  us* Wfoldbf = (us*)alloc((size_t)D * 64 * 2);

  auto cvt = [&](const float* src, us* dst, size_t n) {
    cvt_bf16_kernel<<<dim3((unsigned)(n / 2048)), 256, 0, stream>>>(src, dst, (int)(n / 8));
  };

  // 0) pack inputs to bf16 once (vectorized b128->b128)
  cvt(x, Xbf, (size_t)M * D);
  cvt(Wq, Wqbf, (size_t)D * D);
  cvt(W1, W1bf, (size_t)DFF * D);
  cvt(W2, W2bf, (size_t)D * DFF);

  // 1) Q = x @ Wq^T + PE(s(m), n)
  wmma_gemm_kernel<EP_PE>
      <<<dim3(M / 128, D / 128), 256, 0, stream>>>(Xbf, Wqbf, nullptr, Q, M, D, D);
  // 2) fold Wfc over the 8-fold column tiling induced by the broadcast rows
  fold_wfc_kernel<<<dim3((512 * 64) / 256), 256, 0, stream>>>(Wfc, Wfold);
  // 3) rank-1 attention collapse: softmax of rowsums -> weighted 64-vectors
  attn_reduce_kernel<<<dim3(128), 256, 0, stream>>>(Q, Vr);
  // 4) AO[128,512] = V @ Wfold^T
  cvt(Vr, Vbf, (size_t)128 * 64);
  cvt(Wfold, Wfoldbf, (size_t)D * 64);
  wmma_gemm_kernel<EP_NONE>
      <<<dim3(1, D / 128), 256, 0, stream>>>(Vbf, Wfoldbf, nullptr, AO, 128, D, 64);
  // 5) o1 = LN(x + AO[m>>6]) ; also emit bf16 copy for the FFN GEMM
  ln_kernel<true, true><<<dim3(M / 8), 256, 0, stream>>>(x, AO, g1, be1, O1, O1bf);
  // 6) H = relu(o1 @ W1^T + b1) stored bf16
  wmma_gemm_kernel<EP_RELU_BIAS_BF16>
      <<<dim3(M / 128, DFF / 128), 256, 0, stream>>>(O1bf, W1bf, b1, Hff, M, DFF, D);
  // 7) F = H @ W2^T + b2
  wmma_gemm_kernel<EP_BIAS>
      <<<dim3(M / 128, D / 128), 256, 0, stream>>>(Hff, W2bf, b2, Ff, M, D, DFF);
  // 8) out = LN(o1 + F)
  ln_kernel<false, false>
      <<<dim3(M / 8), 256, 0, stream>>>(O1, Ff, g2, be2, (float*)d_out, nullptr);
}
